// ComponentDetection_51719996178879
// MI455X (gfx1250) — matrix operand order repaired and checked by reading, not compiled, not run on hardware
//
#include <hip/hip_runtime.h>

typedef __attribute__((ext_vector_type(16))) __bf16 v16bf;
typedef __attribute__((ext_vector_type(8)))  float  v8f;

#define NROI 64
#define RH 28
#define RW 28
#define NPIX (RH*RW)          // 784
#define FH 128
#define FW 128
#define CFEAT 64
#define COUT 256

// ---------------------------------------------------------------------------
// ROI align: cnn_feature [2][64][128][128] f32 -> act0 [64][784][64] bf16 NHWC
// ---------------------------------------------------------------------------
__global__ __launch_bounds__(256) void roi_align_kernel(
    const float* __restrict__ feat, const int* __restrict__ act_ind,
    const float* __restrict__ awh, __bf16* __restrict__ out) {
  int roi = blockIdx.x;                 // 0..63
  int b = roi >> 5;                     // batch
  int ind = act_ind[roi];
  float xs = (float)(ind % FW), ys = (float)(ind / FW);
  float aw = awh[roi * 2 + 0], ah = awh[roi * 2 + 1];
  float x1 = xs - aw * 0.5f, y1 = ys - ah * 0.5f;
  float bw = aw / (float)RW, bh = ah / (float)RH;
  const float* img = feat + (size_t)b * CFEAT * FH * FW;

  // each iteration handles one output pixel x one 16-channel quarter
  for (int i = threadIdx.x; i < NPIX * 4; i += blockDim.x) {
    int p = i >> 2, cq = i & 3;
    int py = p / RW, px = p % RW;
    float acc[16];
#pragma unroll
    for (int c = 0; c < 16; ++c) acc[c] = 0.f;
#pragma unroll
    for (int s = 0; s < 4; ++s) {
      int jy = s >> 1, jx = s & 1;
      float fy = (float)py + ((float)jy + 0.5f) * 0.5f;
      float fx = (float)px + ((float)jx + 0.5f) * 0.5f;
      float ysr = fminf(fmaxf(y1 + fy * bh, 0.f), (float)(FH - 1));
      float xsr = fminf(fmaxf(x1 + fx * bw, 0.f), (float)(FW - 1));
      int y0 = (int)floorf(ysr);
      int y1i = min(y0 + 1, FH - 1);
      float wy = ysr - (float)y0;
      int x0 = (int)floorf(xsr);
      int x1i = min(x0 + 1, FW - 1);
      float wx = xsr - (float)x0;
      float w00 = (1.f - wy) * (1.f - wx), w01 = (1.f - wy) * wx;
      float w10 = wy * (1.f - wx), w11 = wy * wx;
#pragma unroll
      for (int c = 0; c < 16; ++c) {
        const float* ch = img + (size_t)(cq * 16 + c) * FH * FW;
        float v = ch[y0 * FW + x0] * w00 + ch[y0 * FW + x1i] * w01 +
                  ch[y1i * FW + x0] * w10 + ch[y1i * FW + x1i] * w11;
        acc[c] += v * 0.25f;
      }
    }
    __bf16* dst = out + ((size_t)roi * NPIX + p) * CFEAT + cq * 16;
#pragma unroll
    for (int c = 0; c < 16; ++c) dst[c] = (__bf16)acc[c];
  }
}

// ---------------------------------------------------------------------------
// Swizzle OIHW f32 weights into per-lane A-fragment layout (bf16):
//   wswz[oct(16)][kb(CIN/32)][khw(9)][lane(32)][elem(16)]
// lane<16 : oc=oct*16+lane,   elems = K {0..15} of the 32-block
// lane>=16: oc=oct*16+lane-16, elems = K {16..31}
// ---------------------------------------------------------------------------
template <int CIN>
__global__ __launch_bounds__(256) void swizzle_weights_kernel(
    const float* __restrict__ w, __bf16* __restrict__ wswz) {
  const int KB = CIN / 32;
  const size_t total = (size_t)16 * KB * 9 * 32 * 16;
  for (size_t i = (size_t)blockIdx.x * blockDim.x + threadIdx.x; i < total;
       i += (size_t)gridDim.x * blockDim.x) {
    int e = (int)(i & 15);
    size_t t = i >> 4;
    int l = (int)(t & 31); t >>= 5;
    int khw = (int)(t % 9); t /= 9;
    int kb = (int)(t % KB); t /= KB;
    int oct = (int)t;
    int oc = oct * 16 + (l & 15);
    int kk;
    if (l < 16) kk = e;
    else        kk = 16 + e;
    int cin = kb * 32 + kk;
    int kh = khw / 3, kw = khw % 3;
    float val = w[(((size_t)oc * CIN + cin) * 3 + kh) * 3 + kw];
    wswz[i] = (__bf16)val;
  }
}

// ---------------------------------------------------------------------------
// 3x3 SAME conv + bias + ReLU, implicit GEMM via V_WMMA_F32_16X16X32_BF16.
// in:  [64][784][CIN] bf16 NHWC     out: [64][784][256] bf16 NHWC
// grid = (28 rows, 64 images), block = 256 (8 waves).
// One block covers a FULL row (two 16-pixel N-tiles) so every A (weight)
// fragment feeds two WMMAs -> halves L2 weight traffic vs a 16-pixel tile.
// Wave w owns oc-tiles {w, w+8}; K staged 32 channels at a time in LDS.
// ---------------------------------------------------------------------------
template <int CIN>
__global__ __launch_bounds__(256) void conv3x3_wmma_kernel(
    const __bf16* __restrict__ in, const __bf16* __restrict__ wswz,
    const float* __restrict__ bias, __bf16* __restrict__ out) {
  const int KB = CIN / 32;
  // halo tile: 3 rows x 34 cols (x = -1 .. 32) x 32 channels, 6.5 KB
  __shared__ __align__(32) __bf16 patch[3][34][32];

  const int tid = threadIdx.x;
  const int lane = tid & 31, wave = tid >> 5;
  const int y = blockIdx.x;
  const int img = blockIdx.y;
  const int oct0 = wave, oct1 = wave + 8;

  v8f c00 = {}, c01 = {}, c10 = {}, c11 = {};

  for (int kb = 0; kb < KB; ++kb) {
    __syncthreads();  // protect previous iteration's LDS reads
    // cooperative load of 3x34 positions x 32 channels (16B chunks)
    for (int i = tid; i < 102 * 4; i += 256) {
      int pos = i >> 2, part = i & 3;
      int r = pos / 34, cc = pos % 34;
      int gy = y + r - 1, gx = cc - 1;
      uint4 val = {0u, 0u, 0u, 0u};
      if ((unsigned)gy < (unsigned)RH && (unsigned)gx < (unsigned)RW) {
        const uint4* src =
            (const uint4*)(in + ((size_t)img * NPIX + gy * RW + gx) * CIN +
                           kb * 32) + part;
        val = *src;
      }
      *((uint4*)(&patch[r][cc][0]) + part) = val;
    }
    __syncthreads();

    const __bf16* wb0 = wswz + ((size_t)(oct0 * KB + kb) * 9) * 512;
    const __bf16* wb1 = wswz + ((size_t)(oct1 * KB + kb) * 9) * 512;
#pragma unroll
    for (int khw = 0; khw < 9; ++khw) {
      int kh = khw / 3, kw = khw % 3;
      int khalf = (lane >> 4) * 16;
      v16bf b0 = *(const v16bf*)&patch[kh][(lane & 15) + kw][khalf];
      v16bf b1 = *(const v16bf*)&patch[kh][(lane & 15) + 16 + kw][khalf];
      v16bf a0 = *(const v16bf*)(wb0 + (size_t)(khw * 32 + lane) * 16);
      v16bf a1 = *(const v16bf*)(wb1 + (size_t)(khw * 32 + lane) * 16);
      c00 = __builtin_amdgcn_wmma_f32_16x16x32_bf16(false, a0, false, b0,
                                                    (short)0, c00, false, false);
      c10 = __builtin_amdgcn_wmma_f32_16x16x32_bf16(false, a1, false, b0,
                                                    (short)0, c10, false, false);
      c01 = __builtin_amdgcn_wmma_f32_16x16x32_bf16(false, a0, false, b1,
                                                    (short)0, c01, false, false);
      c11 = __builtin_amdgcn_wmma_f32_16x16x32_bf16(false, a1, false, b1,
                                                    (short)0, c11, false, false);
    }
  }

  // store: C/D layout -> M = g + 8*(lane>=16), N = lane%16
  int px0 = lane & 15;
  int px1 = 16 + (lane & 15);
  int mhi = (lane >> 4) << 3;
  size_t rowoff = ((size_t)img * NPIX + y * RW) * COUT;
  {
    size_t pixoff = rowoff + (size_t)px0 * COUT;
#pragma unroll
    for (int g = 0; g < 8; ++g) {
      int m = g + mhi;
      int o0 = oct0 * 16 + m, o1 = oct1 * 16 + m;
      out[pixoff + o0] = (__bf16)fmaxf(c00[g] + bias[o0], 0.f);
      out[pixoff + o1] = (__bf16)fmaxf(c10[g] + bias[o1], 0.f);
    }
  }
  if (px1 < RW) {
    size_t pixoff = rowoff + (size_t)px1 * COUT;
#pragma unroll
    for (int g = 0; g < 8; ++g) {
      int m = g + mhi;
      int o0 = oct0 * 16 + m, o1 = oct1 * 16 + m;
      out[pixoff + o0] = (__bf16)fmaxf(c01[g] + bias[o0], 0.f);
      out[pixoff + o1] = (__bf16)fmaxf(c11[g] + bias[o1], 0.f);
    }
  }
}

// ---------------------------------------------------------------------------
// Fold 1x1 heads into the 2x2 deconv weights:
//   fw[ch][dy][dx][ic] = sum_oc headW[ch][oc] * dw[ic][oc][dy][dx]
//   fb[ch] = headB[ch] + sum_oc headW[ch][oc] * deconvB[oc]
// ---------------------------------------------------------------------------
__global__ __launch_bounds__(256) void fold_head_kernel(
    const float* __restrict__ hm_dw, const float* __restrict__ hm_db,
    const float* __restrict__ hm_w, const float* __restrict__ hm_b,
    const float* __restrict__ wh_dw, const float* __restrict__ wh_db,
    const float* __restrict__ wh_w, const float* __restrict__ wh_b,
    float* __restrict__ fw, float* __restrict__ fb) {
  int i = blockIdx.x * blockDim.x + threadIdx.x;
  if (i >= 3 * 4 * 256) return;
  int ic = i & 255, dydx = (i >> 8) & 3, c = i >> 10;
  const float* dw = (c == 0) ? hm_dw : wh_dw;
  const float* hw = (c == 0) ? hm_w : (wh_w + (size_t)(c - 1) * 256);
  float s = 0.f;
  for (int oc = 0; oc < 256; ++oc)
    s += hw[oc] * dw[((size_t)ic * 256 + oc) * 4 + dydx];
  fw[i] = s;
  if (i < 3) {
    const float* db = (i == 0) ? hm_db : wh_db;
    const float* hww = (i == 0) ? hm_w : (wh_w + (size_t)(i - 1) * 256);
    float t = (i == 0) ? hm_b[0] : wh_b[i - 1];
    for (int oc = 0; oc < 256; ++oc) t += hww[oc] * db[oc];
    fb[i] = t;
  }
}

// ---------------------------------------------------------------------------
// Apply folded deconv+head: x [64][784][256] bf16 -> out [64][3][56][56] f32
// out[img][ch][2y+dy][2x+dx] = fb[ch] + sum_ic x[img][y*28+x][ic]*fw[ch][dy][dx][ic]
// ---------------------------------------------------------------------------
__global__ __launch_bounds__(256) void head_kernel(
    const __bf16* __restrict__ x, const float* __restrict__ fw,
    const float* __restrict__ fb, float* __restrict__ out) {
  __shared__ float sfw[3 * 4 * 256];
  for (int j = threadIdx.x; j < 3 * 4 * 256; j += blockDim.x) sfw[j] = fw[j];
  __syncthreads();

  int p = blockIdx.x * blockDim.x + threadIdx.x;
  if (p >= NROI * NPIX) return;
  int img = p / NPIX, pix = p % NPIX;
  int yy = pix / RW, xx = pix % RW;
  const __bf16* xin = x + (size_t)p * COUT;

  float acc[12];
#pragma unroll
  for (int j = 0; j < 12; ++j) acc[j] = 0.f;
  for (int c = 0; c < COUT; ++c) {
    float v = (float)xin[c];
#pragma unroll
    for (int j = 0; j < 12; ++j) acc[j] += v * sfw[j * 256 + c];
  }
#pragma unroll
  for (int ch = 0; ch < 3; ++ch)
#pragma unroll
    for (int dy = 0; dy < 2; ++dy)
#pragma unroll
      for (int dx = 0; dx < 2; ++dx) {
        int j = (ch * 2 + dy) * 2 + dx;
        out[(((size_t)img * 3 + ch) * 56 + (2 * yy + dy)) * 56 +
            (2 * xx + dx)] = acc[j] + fb[ch];
      }
}

// ---------------------------------------------------------------------------
static inline size_t align_up(size_t v, size_t a) { return (v + a - 1) & ~(a - 1); }

extern "C" void kernel_launch(void* const* d_in, const int* in_sizes, int n_in,
                              void* d_out, int out_size, void* d_ws,
                              size_t ws_size, hipStream_t stream) {
  (void)in_sizes; (void)n_in; (void)out_size; (void)ws_size;
  const float* cnn_feature = (const float*)d_in[0];
  const int*   act_ind     = (const int*)d_in[1];
  const float* awh         = (const float*)d_in[2];
  const float* w1 = (const float*)d_in[3];  const float* b1 = (const float*)d_in[4];
  const float* w2 = (const float*)d_in[5];  const float* b2 = (const float*)d_in[6];
  const float* w3 = (const float*)d_in[7];  const float* b3 = (const float*)d_in[8];
  const float* w4 = (const float*)d_in[9];  const float* b4 = (const float*)d_in[10];
  const float* hm_dw = (const float*)d_in[11]; const float* hm_db = (const float*)d_in[12];
  const float* hm_w  = (const float*)d_in[13]; const float* hm_b  = (const float*)d_in[14];
  const float* wh_dw = (const float*)d_in[15]; const float* wh_db = (const float*)d_in[16];
  const float* wh_w  = (const float*)d_in[17]; const float* wh_b  = (const float*)d_in[18];
  float* outp = (float*)d_out;

  // workspace carve-up (all 256B aligned)
  char* base = (char*)d_ws;
  size_t off = 0;
  __bf16* act0 = (__bf16*)(base + off); off = align_up(off + (size_t)NROI * NPIX * CFEAT * 2, 256);
  __bf16* actA = (__bf16*)(base + off); off = align_up(off + (size_t)NROI * NPIX * COUT * 2, 256);
  __bf16* actB = (__bf16*)(base + off); off = align_up(off + (size_t)NROI * NPIX * COUT * 2, 256);
  __bf16* ws1  = (__bf16*)(base + off); off = align_up(off + (size_t)16 * 2 * 9 * 512 * 2, 256);
  __bf16* ws2  = (__bf16*)(base + off); off = align_up(off + (size_t)16 * 8 * 9 * 512 * 2, 256);
  __bf16* ws3  = (__bf16*)(base + off); off = align_up(off + (size_t)16 * 8 * 9 * 512 * 2, 256);
  __bf16* ws4  = (__bf16*)(base + off); off = align_up(off + (size_t)16 * 8 * 9 * 512 * 2, 256);
  float*  fw   = (float*)(base + off);  off = align_up(off + 3 * 4 * 256 * 4, 256);
  float*  fb   = (float*)(base + off);  off = align_up(off + 16, 256);

  // 1. weight prep (bf16 A-fragment swizzle) + head folding
  swizzle_weights_kernel<64><<<288, 256, 0, stream>>>(w1, ws1);
  swizzle_weights_kernel<256><<<1152, 256, 0, stream>>>(w2, ws2);
  swizzle_weights_kernel<256><<<1152, 256, 0, stream>>>(w3, ws3);
  swizzle_weights_kernel<256><<<1152, 256, 0, stream>>>(w4, ws4);
  fold_head_kernel<<<12, 256, 0, stream>>>(hm_dw, hm_db, hm_w, hm_b,
                                           wh_dw, wh_db, wh_w, wh_b, fw, fb);

  // 2. ROI align -> NHWC bf16
  roi_align_kernel<<<NROI, 256, 0, stream>>>(cnn_feature, act_ind, awh, act0);

  // 3. four 3x3 convs (WMMA implicit GEMM, bias+ReLU fused)
  dim3 cgrid(RH, NROI);
  conv3x3_wmma_kernel<64><<<cgrid, 256, 0, stream>>>(act0, ws1, b1, actA);
  conv3x3_wmma_kernel<256><<<cgrid, 256, 0, stream>>>(actA, ws2, b2, actB);
  conv3x3_wmma_kernel<256><<<cgrid, 256, 0, stream>>>(actB, ws3, b3, actA);
  conv3x3_wmma_kernel<256><<<cgrid, 256, 0, stream>>>(actA, ws4, b4, actB);

  // 4. folded deconv+head -> f32 output [64][3][56][56]
  head_kernel<<<(NROI * NPIX + 255) / 256, 256, 0, stream>>>(actB, fw, fb, outp);
}
